// Spiking_Net_16552803958867
// MI455X (gfx1250) — matrix operand order repaired and checked by reading, not compiled, not run on hardware
//
#include <hip/hip_runtime.h>
#include <hip/hip_bf16.h>
#include <stddef.h>
#include <stdint.h>

// ---------------------------------------------------------------------------
// CDNA5 (gfx1250) WMMA types
// ---------------------------------------------------------------------------
typedef __attribute__((ext_vector_type(16))) __bf16 v16bf;
typedef __attribute__((ext_vector_type(8)))  __bf16 v8bf;
typedef __attribute__((ext_vector_type(8)))  float  v8f;

#define TSTEPS 50
#define BATCH  256
#define D0     784
#define D1     2048
#define D2     2048
#define D3     10
#define BETA   0.9f
#define THR    1.0f

#define NROWS  (TSTEPS * BATCH)          // 12800 rows of the batched GEMM
#define L1ELEM (BATCH * D1)              // 524288

// ---------------------------------------------------------------------------
// Layer 1: C1[256][2048] = data[256][784] @ W1^T + b1   (exact fp32, once)
// ---------------------------------------------------------------------------
__global__ __launch_bounds__(256)
void k_c1_gemm(const float* __restrict__ X, const float* __restrict__ W,
               const float* __restrict__ bias, float* __restrict__ C)
{
    __shared__ float sX[64][17];
    __shared__ float sW[64][17];
    const int tx = threadIdx.x & 15, ty = threadIdx.x >> 4;
    const int bj = blockIdx.x * 64;   // output-neuron tile
    const int bb = blockIdx.y * 64;   // batch tile
    float acc[4][4];
    for (int i = 0; i < 4; i++) for (int j = 0; j < 4; j++) acc[i][j] = 0.f;

    for (int k0 = 0; k0 < D0; k0 += 16) {
        for (int i = 0; i < 4; i++) {
            int t = threadIdx.x + i * 256;      // 0..1023 covers 64x16
            int r = t >> 4, c = t & 15;
            sX[r][c] = X[(size_t)(bb + r) * D0 + k0 + c];
            sW[r][c] = W[(size_t)(bj + r) * D0 + k0 + c];
        }
        __syncthreads();
        for (int kk = 0; kk < 16; kk++) {
            float xr[4], wr[4];
            for (int i = 0; i < 4; i++) xr[i] = sX[ty * 4 + i][kk];
            for (int i = 0; i < 4; i++) wr[i] = sW[tx * 4 + i][kk];
            for (int i = 0; i < 4; i++)
                for (int j = 0; j < 4; j++) acc[i][j] += xr[i] * wr[j];
        }
        __syncthreads();
    }
    for (int i = 0; i < 4; i++) {
        int b = bb + ty * 4 + i;
        for (int j = 0; j < 4; j++) {
            int n = bj + tx * 4 + j;
            C[(size_t)b * D1 + n] = acc[i][j] + bias[n];
        }
    }
}

// ---------------------------------------------------------------------------
// LIF scan, layer 1: c1 constant over time -> emit whole spike train in bf16.
// ---------------------------------------------------------------------------
__global__ void k_lif1(const float* __restrict__ C1, __bf16* __restrict__ S1)
{
    const int idx = blockIdx.x * blockDim.x + threadIdx.x;   // b*2048+j
    const float c = C1[idx];
    float m = 0.f;
    for (int t = 0; t < TSTEPS; t++) {
        m = BETA * m + c;
        float s = (m > THR) ? 1.0f : 0.0f;
        m -= s * THR;
        S1[(size_t)t * L1ELEM + idx] = (__bf16)s;
    }
}

// ---------------------------------------------------------------------------
// Split W2 into bf16 hi + bf16 lo (hi+lo ~ 16 mantissa bits of fp32).
// ---------------------------------------------------------------------------
__global__ void k_w2split(const float* __restrict__ W,
                          __bf16* __restrict__ hi, __bf16* __restrict__ lo)
{
    const int idx = blockIdx.x * blockDim.x + threadIdx.x;
    float w = W[idx];
    __bf16 h = (__bf16)w;
    hi[idx] = h;
    lo[idx] = (__bf16)(w - (float)h);
}

// ---------------------------------------------------------------------------
// Dominant batched GEMM:
//   C[12800][2048] = S1 @ (W2hi+W2lo)^T + b2
// Block 128(M) x 64(N), K-step 32, 8 waves. Double-buffered LDS fed by
// GLOBAL_LOAD_ASYNC_TO_LDS_B128 (ASYNCcnt), compute on v_wmma_f32_16x16x32_bf16.
// Fragments are loaded into distinct registers first, then 8 WMMAs issue
// back-to-back (same-accumulator reuse distance 4).
// ---------------------------------------------------------------------------
#define GBM 128
#define GBN 64
#define GBK 32
#define LDS_STRIDE 40   // padded row stride (halves); keeps 16B alignment

__global__ __launch_bounds__(256)
void k_gemm_wmma(const __bf16* __restrict__ A,    // [M][K]  spikes
                 const __bf16* __restrict__ Bhi,  // [N][K]  W2 hi
                 const __bf16* __restrict__ Blo,  // [N][K]  W2 lo
                 const float*  __restrict__ bias, // [N]
                 float* __restrict__ C,           // [M][N]
                 int M, int N, int K)
{
    __shared__ __bf16 sA [2][GBM * LDS_STRIDE];
    __shared__ __bf16 sBh[2][GBN * LDS_STRIDE];
    __shared__ __bf16 sBl[2][GBN * LDS_STRIDE];

    const int tid  = threadIdx.x;
    const int wave = tid >> 5;
    const int lane = tid & 31;
    const int m0 = blockIdx.x * GBM;
    const int n0 = blockIdx.y * GBN;

    v8f acc[4];
    for (int i = 0; i < 4; i++)
        for (int e = 0; e < 8; e++) acc[i][e] = 0.f;

    // cooperative async-copy task assignment (32B per thread per tile)
    const int arow = tid >> 1;               // 0..127
    const int aseg = tid & 1;                // two 16-half segments per row
    const int bsel = tid >> 7;               // 0 -> hi, 1 -> lo
    const int brow = (tid & 127) >> 1;       // 0..63
    const int bseg = tid & 1;

    const __bf16* gA = A + (size_t)(m0 + arow) * K + aseg * 16;
    const __bf16* gB = (bsel ? Blo : Bhi) + (size_t)(n0 + brow) * K + bseg * 16;

    // Per-stage async issue: 4 x GLOBAL_LOAD_ASYNC_TO_LDS_B128 per thread.
    // offset: applies to BOTH the global and LDS address.
    auto issue = [&](int kk, int buf) {
        unsigned la = (unsigned)(uintptr_t)&sA[buf][arow * LDS_STRIDE + aseg * 16];
        unsigned long long ga = (unsigned long long)(uintptr_t)(gA + kk);
        asm volatile("global_load_async_to_lds_b128 %0, %1, off"
                     :: "v"(la), "v"(ga) : "memory");
        asm volatile("global_load_async_to_lds_b128 %0, %1, off offset:16"
                     :: "v"(la), "v"(ga) : "memory");
        __bf16* db = bsel ? sBl[buf] : sBh[buf];
        unsigned lb = (unsigned)(uintptr_t)&db[brow * LDS_STRIDE + bseg * 16];
        unsigned long long gb = (unsigned long long)(uintptr_t)(gB + kk);
        asm volatile("global_load_async_to_lds_b128 %0, %1, off"
                     :: "v"(lb), "v"(gb) : "memory");
        asm volatile("global_load_async_to_lds_b128 %0, %1, off offset:16"
                     :: "v"(lb), "v"(gb) : "memory");
    };

    // per-wave fragment addressing (ISA 7.12.2 bf16 layouts)
    const int a_r  = wave * 16 + (lane & 15);        // A row within block tile
    const int a_k  = (lane >> 4) * 8;                // K sub-offset {0,8}
    const int b_n  = (lane & 15);                    // B col within subtile
    const int b_k  = (lane >> 4) * 8;                // K sub-offset {0,8}

    // Compute one K-stage out of LDS buffer `buf`:
    // load A frag + all 8 B frags into distinct registers, then 8 WMMAs.
    auto compute = [&](int buf) {
        const __bf16* pa = &sA[buf][a_r * LDS_STRIDE + a_k];
        v8bf alo = *reinterpret_cast<const v8bf*>(pa);
        v8bf ahi = *reinterpret_cast<const v8bf*>(pa + 16);
        v16bf afrag = __builtin_shufflevector(alo, ahi,
            0,1,2,3,4,5,6,7,8,9,10,11,12,13,14,15);

        v16bf bh[4], bl[4];
#pragma unroll
        for (int nt = 0; nt < 4; nt++) {
            const int boff = (nt * 16 + b_n) * LDS_STRIDE + b_k;
            const __bf16* ph = &sBh[buf][boff];
            v8bf h0 = *reinterpret_cast<const v8bf*>(ph);
            v8bf h1 = *reinterpret_cast<const v8bf*>(ph + 16);
            bh[nt] = __builtin_shufflevector(h0, h1,
                0,1,2,3,4,5,6,7,8,9,10,11,12,13,14,15);
            const __bf16* pl = &sBl[buf][boff];
            v8bf l0 = *reinterpret_cast<const v8bf*>(pl);
            v8bf l1 = *reinterpret_cast<const v8bf*>(pl + 16);
            bl[nt] = __builtin_shufflevector(l0, l1,
                0,1,2,3,4,5,6,7,8,9,10,11,12,13,14,15);
        }
#pragma unroll
        for (int nt = 0; nt < 4; nt++)
            acc[nt] = __builtin_amdgcn_wmma_f32_16x16x32_bf16(
                false, afrag, false, bh[nt], (short)0, acc[nt], false, false);
#pragma unroll
        for (int nt = 0; nt < 4; nt++)
            acc[nt] = __builtin_amdgcn_wmma_f32_16x16x32_bf16(
                false, afrag, false, bl[nt], (short)0, acc[nt], false, false);
    };

    const int niter = K / GBK;
    issue(0, 0);

    // Steady state: prefetch stage it+1 while computing stage it.
    for (int it = 0; it < niter - 1; ++it) {
        const int buf = it & 1;
        issue((it + 1) * GBK, buf ^ 1);
        // previous stage's 4 async ops are older -> done once <=4 remain
        asm volatile("s_wait_asynccnt 0x4" ::: "memory");
        __syncthreads();            // publish stage `it` LDS tile
        compute(buf);
        __syncthreads();            // reads of buf done before it+2 overwrite
    }

    // Peeled last stage (no further prefetch).
    asm volatile("s_wait_asynccnt 0x0" ::: "memory");
    __syncthreads();
    compute((niter - 1) & 1);

    // Epilogue: C/D layout -> lanes 0-15 give M=r, lanes 16-31 give M=8+r.
    const int mbase = m0 + wave * 16 + ((lane >> 4) ? 8 : 0);
    for (int nt = 0; nt < 4; nt++) {
        const int n = n0 + nt * 16 + (lane & 15);
        const float bv = bias[n];
        for (int r = 0; r < 8; r++)
            C[(size_t)(mbase + r) * N + n] = acc[nt][r] + bv;
    }
}

// ---------------------------------------------------------------------------
// LIF scan for layer 2.
// ---------------------------------------------------------------------------
__global__ void k_lif2(const float* __restrict__ C2, __bf16* __restrict__ S2)
{
    const int idx = blockIdx.x * blockDim.x + threadIdx.x;   // b*2048+j
    float m = 0.f;
    for (int t = 0; t < TSTEPS; t++) {
        m = BETA * m + C2[(size_t)t * L1ELEM + idx];
        float s = (m > THR) ? 1.0f : 0.0f;
        m -= s * THR;
        S2[(size_t)t * L1ELEM + idx] = (__bf16)s;
    }
}

// ---------------------------------------------------------------------------
// Layer 3 currents: C3[row][k] = S2[row][:] . W3[k][:] + b3[k], row = t*256+b.
// ---------------------------------------------------------------------------
__global__ __launch_bounds__(256)
void k_c3(const __bf16* __restrict__ S2, const float* __restrict__ W3,
          const float* __restrict__ b3, float* __restrict__ C3)
{
    __shared__ __bf16 sRow[D2];
    __shared__ float  sRed[8];
    const int row = blockIdx.x;
    const int tid = threadIdx.x;

    // 2048 bf16 = 256 x uint4
    reinterpret_cast<uint4*>(sRow)[tid] =
        reinterpret_cast<const uint4*>(S2 + (size_t)row * D2)[tid];
    __syncthreads();

    for (int k = 0; k < D3; k++) {
        float p = 0.f;
        for (int j = tid; j < D2; j += 256)
            p += (float)sRow[j] * W3[(size_t)k * D2 + j];
        for (int off = 16; off > 0; off >>= 1)
            p += __shfl_down(p, off, 32);
        if ((tid & 31) == 0) sRed[tid >> 5] = p;
        __syncthreads();
        if (tid == 0) {
            float sum = b3[k];
            for (int w = 0; w < 8; w++) sum += sRed[w];
            C3[(size_t)row * D3 + k] = sum;
        }
        __syncthreads();
    }
}

// ---------------------------------------------------------------------------
// LIF scan for layer 3 + final output spikes [T,B,10] fp32.
// ---------------------------------------------------------------------------
__global__ void k_lif3(const float* __restrict__ C3, float* __restrict__ out)
{
    const int idx = blockIdx.x * blockDim.x + threadIdx.x;   // b*10+k
    if (idx >= BATCH * D3) return;
    float m = 0.f;
    for (int t = 0; t < TSTEPS; t++) {
        m = BETA * m + C3[(size_t)t * (BATCH * D3) + idx];
        float s = (m > THR) ? 1.0f : 0.0f;
        m -= s * THR;
        out[(size_t)t * (BATCH * D3) + idx] = s;
    }
}

// ---------------------------------------------------------------------------
// Launcher
// ---------------------------------------------------------------------------
extern "C" void kernel_launch(void* const* d_in, const int* in_sizes, int n_in,
                              void* d_out, int out_size, void* d_ws, size_t ws_size,
                              hipStream_t stream)
{
    const float* data = (const float*)d_in[0];   // [256,784]
    const float* W1   = (const float*)d_in[1];   // [2048,784]
    const float* b1   = (const float*)d_in[2];   // [2048]
    const float* W2   = (const float*)d_in[3];   // [2048,2048]
    const float* b2   = (const float*)d_in[4];   // [2048]
    const float* W3   = (const float*)d_in[5];   // [10,2048]
    const float* b3   = (const float*)d_in[6];   // [10]
    float* out = (float*)d_out;                  // [50,256,10]

    char* ws = (char*)d_ws;
    size_t off = 0;
    float*  C1   = (float*) (ws + off); off += (size_t)BATCH * D1 * 4;        // 2 MB
    __bf16* S1   = (__bf16*)(ws + off); off += (size_t)NROWS * D1 * 2;        // 52 MB
    __bf16* W2hi = (__bf16*)(ws + off); off += (size_t)D2 * D1 * 2;           // 8 MB
    __bf16* W2lo = (__bf16*)(ws + off); off += (size_t)D2 * D1 * 2;           // 8 MB
    float*  C2   = (float*) (ws + off); off += (size_t)NROWS * D2 * 4;        // 105 MB
    __bf16* S2   = (__bf16*)(ws + off); off += (size_t)NROWS * D2 * 2;        // 52 MB
    float*  C3   = (float*) (ws + off); off += (size_t)NROWS * D3 * 4;        // 0.5 MB

    // Layer 1 (once): exact fp32 GEMM, then full 50-step spike train.
    k_c1_gemm<<<dim3(D1 / 64, BATCH / 64), 256, 0, stream>>>(data, W1, b1, C1);
    k_lif1<<<L1ELEM / 256, 256, 0, stream>>>(C1, S1);

    // Weight split for near-fp32 bf16 WMMA.
    k_w2split<<<(D2 * D1) / 256, 256, 0, stream>>>(W2, W2hi, W2lo);

    // The dominant batched GEMM on v_wmma_f32_16x16x32_bf16 with async-LDS feed.
    k_gemm_wmma<<<dim3(NROWS / GBM, D2 / GBN), 256, 0, stream>>>(
        S1, W2hi, W2lo, b2, C2, NROWS, D2, D1);

    k_lif2<<<L1ELEM / 256, 256, 0, stream>>>(C2, S2);

    // Layer 3 currents + final LIF.
    k_c3<<<NROWS, 256, 0, stream>>>(S2, W3, b3, C3);
    k_lif3<<<(BATCH * D3 + 255) / 256, 256, 0, stream>>>(C3, out);
}
